// MoE_42614665511161
// MI455X (gfx1250) — matrix operand order repaired and checked by reading, not compiled, not run on hardware
//
#include <hip/hip_runtime.h>
#include <hip/hip_bf16.h>

// ---------------------------------------------------------------------------
// MoE (8 experts, top-2, SwiGLU) + shared expert for MI455X / gfx1250.
// Strategy: routed (sparse) expert compute with f16 WMMA (f32 accumulate).
// ---------------------------------------------------------------------------

typedef __attribute__((ext_vector_type(16))) _Float16 v16h;
typedef __attribute__((ext_vector_type(8)))  float    v8f;

#define T_TOK   4096
#define DIM_    2048
#define INTER_  1408
#define NEXP_   8
#define SHI_    2816   // N_SHARED * INTER

static __device__ __forceinline__ float sigmoidf_(float x) {
    return 1.0f / (1.0f + __expf(-x));
}

// ---------------------------------------------------------------------------
// counter init
// ---------------------------------------------------------------------------
__global__ void moe_init_kernel(int* __restrict__ counts) {
    if (threadIdx.x < NEXP_) counts[threadIdx.x] = 0;
}

// ---------------------------------------------------------------------------
// fp32 -> f16 conversion of X (done once; X is reused by 3 GEMMs)
// ---------------------------------------------------------------------------
__global__ void moe_cvt_kernel(const float* __restrict__ x,
                               _Float16* __restrict__ y, int n) {
    int i = (blockIdx.x * blockDim.x + threadIdx.x) * 4;
    if (i + 3 < n) {
        float4 v = *reinterpret_cast<const float4*>(x + i);
        union { _Float16 h[4]; uint2 u; } o;
        o.h[0] = (_Float16)v.x; o.h[1] = (_Float16)v.y;
        o.h[2] = (_Float16)v.z; o.h[3] = (_Float16)v.w;
        *reinterpret_cast<uint2*>(y + i) = o.u;
    }
}

// ---------------------------------------------------------------------------
// Router: sigmoid scores, biased top-2 (lowest index on ties, like lax.top_k),
// normalized weights; builds compacted per-expert token lists with atomics.
// ---------------------------------------------------------------------------
__global__ void moe_router_kernel(const float* __restrict__ X,
                                  const float* __restrict__ GW,
                                  const float* __restrict__ bias,
                                  int* __restrict__ counts,
                                  int* __restrict__ tokenList,
                                  float* __restrict__ gateW) {
    int t = blockIdx.x * blockDim.x + threadIdx.x;
    if (t >= T_TOK) return;
    const float* x = X + (size_t)t * DIM_;
    float acc[NEXP_];
#pragma unroll
    for (int e = 0; e < NEXP_; ++e) acc[e] = 0.0f;
    for (int d = 0; d < DIM_; ++d) {
        float xv = x[d];
#pragma unroll
        for (int e = 0; e < NEXP_; ++e) acc[e] += xv * GW[e * DIM_ + d];
    }
    float sc[NEXP_], sel[NEXP_];
#pragma unroll
    for (int e = 0; e < NEXP_; ++e) {
        sc[e]  = sigmoidf_(acc[e]);
        sel[e] = sc[e] + bias[e];
    }
    int i0 = 0;
#pragma unroll
    for (int e = 1; e < NEXP_; ++e) if (sel[e] > sel[i0]) i0 = e;
    int i1 = -1;
#pragma unroll
    for (int e = 0; e < NEXP_; ++e) {
        if (e == i0) continue;
        if (i1 < 0 || sel[e] > sel[i1]) i1 = e;
    }
    float w0 = sc[i0], w1 = sc[i1];
    float denom = fmaxf(w0 + w1, 1e-12f);
    w0 /= denom; w1 /= denom;
    int s0 = atomicAdd(&counts[i0], 1);
    tokenList[i0 * T_TOK + s0] = t;  gateW[i0 * T_TOK + s0] = w0;
    int s1 = atomicAdd(&counts[i1], 1);
    tokenList[i1 * T_TOK + s1] = t;  gateW[i1 * T_TOK + s1] = w1;
}

// ---------------------------------------------------------------------------
// Tiled GEMM using v_wmma_f32_16x16x32_f16.
//   block = 256 threads = 8 waves, tile 128(M) x 64(N), K-step 32.
//   Each wave: 16(M) x 64(N) -> 4 accumulators (8 in dual-GEMM mode).
// MODE 0: D1 = A*B1, D3 = A*B3, Out = f16( silu(D1)*D3 )  (SwiGLU hidden)
// MODE 1: OutF[tok] += gate * (A*B1)  (scatter w/ atomicAdd)
// MODE 2: OutF = A*B1                 (plain store)
// KT: B stored [k][n] (k-major). !KT: B stored [n][k].
// ---------------------------------------------------------------------------
template <int MODE, bool KT>
__global__ void __launch_bounds__(256)
moe_gemm_kernel(const _Float16* __restrict__ A, size_t aStrideZ, int lda,
                const float* __restrict__ B1, const float* __restrict__ B3,
                size_t bStrideZ, int ldb, int Ktot,
                const int* __restrict__ counts, int mTotal,
                const int* __restrict__ tokenList, size_t tlStrideZ,
                const float* __restrict__ gateW,
                _Float16* __restrict__ OutH, size_t ohStrideZ, int ldoh,
                float* __restrict__ OutF, int ldof) {
    const int z   = blockIdx.z;
    const int cnt = counts ? counts[z] : mTotal;
    const int mBase = blockIdx.y * 128;
    if (mBase >= cnt) return;                 // uniform early-exit (worst-case grid)
    const int nBase = blockIdx.x * 64;

    const _Float16* Az  = A  + (size_t)z * aStrideZ;
    const float*    B1z = B1 + (size_t)z * bStrideZ;
    const float*    B3z = B3 ? (B3 + (size_t)z * bStrideZ) : nullptr;
    const int*      tl  = tokenList ? (tokenList + (size_t)z * tlStrideZ) : nullptr;

    __shared__ _Float16 sA [128 * 48];        // padded stride: 48 halves = 96 B
    __shared__ _Float16 sB1[ 64 * 48];        // B tile stored transposed: [n][k]
    __shared__ _Float16 sB3[ 64 * 48];

    const int tid  = threadIdx.x;
    const int lane = tid & 31;
    const int wave = tid >> 5;
    const int nl   = lane & 15;
    const int koff = (lane < 16) ? 0 : 8;     // B fragment K offset
    const int aoff = (lane < 16) ? 0 : 8;     // A fragment K offset
    const int rowAdd = (lane < 16) ? 0 : 8;   // C/D row offset

    // ---- A staging assignment: 256 threads cover 128 rows x 32 k (16 h each)
    const int arow_l = tid >> 1;              // 0..127
    const int akpart = (tid & 1) * 16;        // 0 / 16
    const int aslot  = mBase + arow_l;
    const bool avalid = aslot < cnt;
    const _Float16* arow_ptr = nullptr;
    if (avalid) {
        int arow_g = (MODE == 0 && tl) ? tl[aslot] : aslot;
        arow_ptr = Az + (size_t)arow_g * lda;
    }

    // ---- B staging assignments
    // KT: thread owns k-rows {kq, kq+1} x 4 n-columns -> ushort2 (b32) LDS stores
    const int kq     = (tid >> 4) * 2;        // 0,2,..,30
    const int nqpart = (tid & 15) * 4;        // 0..60
    // !KT: thread owns 1 n-row x 8 contiguous k -> single b128 LDS store
    const int nrow   = tid >> 2;              // 0..63
    const int kpart  = (tid & 3) * 8;         // 0..24

    v8f acc1[4], acc3[4];
#pragma unroll
    for (int t = 0; t < 4; ++t)
#pragma unroll
        for (int j = 0; j < 8; ++j) { acc1[t][j] = 0.0f; acc3[t][j] = 0.0f; }

    union F16 { v16h v; uint4 q[2]; };

    for (int kk = 0; kk < Ktot; kk += 32) {
        // ---- prefetch next K-step (overlaps HBM fetch with this step's WMMA) ----
        const int kn = kk + 32;
        if (kn < Ktot) {
            if (avalid) __builtin_prefetch(arow_ptr + kn + akpart, 0, 3);
            if (KT) {
                const float* p0 = B1z + (size_t)(kn + kq) * ldb + nBase + nqpart;
                __builtin_prefetch(p0, 0, 3);
                __builtin_prefetch(p0 + ldb, 0, 3);
                if constexpr (MODE == 0) {
                    const float* p3 = B3z + (size_t)(kn + kq) * ldb + nBase + nqpart;
                    __builtin_prefetch(p3, 0, 3);
                    __builtin_prefetch(p3 + ldb, 0, 3);
                }
            } else {
                __builtin_prefetch(B1z + (size_t)(nBase + nrow) * ldb + kn + kpart, 0, 3);
                if constexpr (MODE == 0)
                    __builtin_prefetch(B3z + (size_t)(nBase + nrow) * ldb + kn + kpart, 0, 3);
            }
        }

        // ---- stage A (f16, 32 B vector loads -> b128 LDS stores) ----
        {
            uint4 q0 = make_uint4(0, 0, 0, 0), q1 = q0;
            if (avalid) {
                const uint4* p = reinterpret_cast<const uint4*>(arow_ptr + kk + akpart);
                q0 = p[0]; q1 = p[1];
            }
            uint4* d = reinterpret_cast<uint4*>(&sA[arow_l * 48 + akpart]);
            d[0] = q0; d[1] = q1;
        }
        // ---- stage B (fp32 -> f16, transposed into [n][k]) ----
        if (KT) {
            // B is [k][n]: read a float4 from rows kq and kq+1, pack (k,k+1)
            // pairs and emit b32 LDS stores into the transposed tile.
            {
                const float* s0 = B1z + (size_t)(kk + kq) * ldb + nBase + nqpart;
                float4 f0 = *reinterpret_cast<const float4*>(s0);
                float4 f1 = *reinterpret_cast<const float4*>(s0 + ldb);
                float a0[4] = {f0.x, f0.y, f0.z, f0.w};
                float a1[4] = {f1.x, f1.y, f1.z, f1.w};
#pragma unroll
                for (int j = 0; j < 4; ++j) {
                    union { _Float16 h[2]; unsigned u; } pk;
                    pk.h[0] = (_Float16)a0[j];
                    pk.h[1] = (_Float16)a1[j];
                    *reinterpret_cast<unsigned*>(&sB1[(nqpart + j) * 48 + kq]) = pk.u;
                }
            }
            if constexpr (MODE == 0) {
                const float* s0 = B3z + (size_t)(kk + kq) * ldb + nBase + nqpart;
                float4 f0 = *reinterpret_cast<const float4*>(s0);
                float4 f1 = *reinterpret_cast<const float4*>(s0 + ldb);
                float a0[4] = {f0.x, f0.y, f0.z, f0.w};
                float a1[4] = {f1.x, f1.y, f1.z, f1.w};
#pragma unroll
                for (int j = 0; j < 4; ++j) {
                    union { _Float16 h[2]; unsigned u; } pk;
                    pk.h[0] = (_Float16)a0[j];
                    pk.h[1] = (_Float16)a1[j];
                    *reinterpret_cast<unsigned*>(&sB3[(nqpart + j) * 48 + kq]) = pk.u;
                }
            }
        } else {
            // B is [n][k]: read contiguous k, pack 8 halves, single b128 store
            {
                const float* s = B1z + (size_t)(nBase + nrow) * ldb + kk + kpart;
                float4 f0 = *reinterpret_cast<const float4*>(s);
                float4 f1 = *reinterpret_cast<const float4*>(s + 4);
                union { _Float16 h[8]; uint4 q; } pk;
                pk.h[0] = (_Float16)f0.x; pk.h[1] = (_Float16)f0.y;
                pk.h[2] = (_Float16)f0.z; pk.h[3] = (_Float16)f0.w;
                pk.h[4] = (_Float16)f1.x; pk.h[5] = (_Float16)f1.y;
                pk.h[6] = (_Float16)f1.z; pk.h[7] = (_Float16)f1.w;
                *reinterpret_cast<uint4*>(&sB1[nrow * 48 + kpart]) = pk.q;
            }
            if constexpr (MODE == 0) {
                const float* s = B3z + (size_t)(nBase + nrow) * ldb + kk + kpart;
                float4 f0 = *reinterpret_cast<const float4*>(s);
                float4 f1 = *reinterpret_cast<const float4*>(s + 4);
                union { _Float16 h[8]; uint4 q; } pk;
                pk.h[0] = (_Float16)f0.x; pk.h[1] = (_Float16)f0.y;
                pk.h[2] = (_Float16)f0.z; pk.h[3] = (_Float16)f0.w;
                pk.h[4] = (_Float16)f1.x; pk.h[5] = (_Float16)f1.y;
                pk.h[6] = (_Float16)f1.z; pk.h[7] = (_Float16)f1.w;
                *reinterpret_cast<uint4*>(&sB3[nrow * 48 + kpart]) = pk.q;
            }
        }
        __syncthreads();

        // ---- A fragment: lane holds row (lane&15), K = {aoff..aoff+7, aoff+16..aoff+23}
        F16 fa;
        {
            const int r = wave * 16 + nl;
            const uint4* p = reinterpret_cast<const uint4*>(&sA[r * 48 + aoff]);
            fa.q[0] = p[0];
            fa.q[1] = p[2];   // +16 halves = +32 B
        }
#pragma unroll
        for (int t = 0; t < 4; ++t) {
            // B fragment: lane holds column nl, K = {koff..koff+7, koff+16..koff+23}
            F16 fb;
            const uint4* p = reinterpret_cast<const uint4*>(&sB1[(t * 16 + nl) * 48 + koff]);
            fb.q[0] = p[0]; fb.q[1] = p[2];
            acc1[t] = __builtin_amdgcn_wmma_f32_16x16x32_f16(
                false, fa.v, false, fb.v, (short)0, acc1[t], false, false);
            if constexpr (MODE == 0) {
                F16 fc;
                const uint4* p3 = reinterpret_cast<const uint4*>(&sB3[(t * 16 + nl) * 48 + koff]);
                fc.q[0] = p3[0]; fc.q[1] = p3[2];
                acc3[t] = __builtin_amdgcn_wmma_f32_16x16x32_f16(
                    false, fa.v, false, fc.v, (short)0, acc3[t], false, false);
            }
        }
        __syncthreads();
    }

    // ---- epilogue: C/D layout — VGPR v: rows {v | v+8}, N = lane&15 ----
    if constexpr (MODE == 0) {
#pragma unroll
        for (int v = 0; v < 8; ++v) {
            int slot = mBase + wave * 16 + v + rowAdd;
            if (slot < cnt) {
                _Float16* dst = OutH + (size_t)z * ohStrideZ + (size_t)slot * ldoh + nBase + nl;
#pragma unroll
                for (int t = 0; t < 4; ++t) {
                    float h1 = acc1[t][v], h3 = acc3[t][v];
                    dst[t * 16] = (_Float16)(h1 * sigmoidf_(h1) * h3);   // silu(h1)*h3
                }
            }
        }
    } else if constexpr (MODE == 1) {
        const float* gz = gateW + (size_t)z * tlStrideZ;
#pragma unroll
        for (int v = 0; v < 8; ++v) {
            int slot = mBase + wave * 16 + v + rowAdd;
            if (slot < cnt) {
                int tok = tl[slot];
                float scale = gz[slot];
                float* dst = OutF + (size_t)tok * ldof + nBase + nl;
#pragma unroll
                for (int t = 0; t < 4; ++t)
                    atomicAdd(dst + t * 16, acc1[t][v] * scale);
            }
        }
    } else {
#pragma unroll
        for (int v = 0; v < 8; ++v) {
            int row = mBase + wave * 16 + v + rowAdd;
            if (row < cnt) {
                float* dst = OutF + (size_t)row * ldof + nBase + nl;
#pragma unroll
                for (int t = 0; t < 4; ++t) dst[t * 16] = acc1[t][v];
            }
        }
    }
}

// ---------------------------------------------------------------------------
// launch
// ---------------------------------------------------------------------------
extern "C" void kernel_launch(void* const* d_in, const int* in_sizes, int n_in,
                              void* d_out, int out_size, void* d_ws, size_t ws_size,
                              hipStream_t stream) {
    (void)in_sizes; (void)n_in; (void)out_size; (void)ws_size;
    const float* X    = (const float*)d_in[0];
    const float* GW   = (const float*)d_in[1];
    const float* BIAS = (const float*)d_in[2];
    const float* W1   = (const float*)d_in[3];
    const float* W3   = (const float*)d_in[4];
    const float* W2   = (const float*)d_in[5];
    const float* SW1  = (const float*)d_in[6];
    const float* SW3  = (const float*)d_in[7];
    const float* SW2  = (const float*)d_in[8];
    float* OUT = (float*)d_out;

    // workspace layout (all chunks 256 B-aligned)
    char* base = (char*)d_ws;
    size_t off = 0;
    int*      counts    = (int*)(base + off);      off += 256;
    int*      tokenList = (int*)(base + off);      off += (size_t)NEXP_ * T_TOK * 4;
    float*    gateW     = (float*)(base + off);    off += (size_t)NEXP_ * T_TOK * 4;
    _Float16* Xh        = (_Float16*)(base + off); off += (size_t)T_TOK * DIM_ * 2;
    _Float16* H         = (_Float16*)(base + off); off += (size_t)NEXP_ * T_TOK * INTER_ * 2;
    _Float16* Hs        = (_Float16*)(base + off); off += (size_t)T_TOK * SHI_ * 2;
    // total ~132 MB

    // 1) zero per-expert counters
    moe_init_kernel<<<1, 32, 0, stream>>>(counts);

    // 2) X -> f16
    moe_cvt_kernel<<<(T_TOK * DIM_ / 4) / 256, 256, 0, stream>>>(X, Xh, T_TOK * DIM_);

    // 3) router + token-list build
    moe_router_kernel<<<T_TOK / 256, 256, 0, stream>>>(X, GW, BIAS, counts, tokenList, gateW);

    // 4) expert hidden: H[e] = silu(Xg*W1[e]) * (Xg*W3[e])   (B layout [k][n])
    moe_gemm_kernel<0, true><<<dim3(INTER_ / 64, T_TOK / 128, NEXP_), 256, 0, stream>>>(
        Xh, 0, DIM_,
        W1, W3, (size_t)DIM_ * INTER_, INTER_, DIM_,
        counts, T_TOK, tokenList, T_TOK, nullptr,
        H, (size_t)T_TOK * INTER_, INTER_, nullptr, 0);

    // 5) shared hidden: Hs = silu(X*sw1^T) * (X*sw3^T)   (B layout [n][k])
    moe_gemm_kernel<0, false><<<dim3(SHI_ / 64, T_TOK / 128, 1), 256, 0, stream>>>(
        Xh, 0, DIM_,
        SW1, SW3, 0, DIM_, DIM_,
        nullptr, T_TOK, nullptr, 0, nullptr,
        Hs, 0, SHI_, nullptr, 0);

    // 6) out = Hs * sw2^T  (plain store; establishes the base value of OUT)
    moe_gemm_kernel<2, false><<<dim3(DIM_ / 64, T_TOK / 128, 1), 256, 0, stream>>>(
        Hs, 0, SHI_,
        SW2, nullptr, 0, SHI_, SHI_,
        nullptr, T_TOK, nullptr, 0, nullptr,
        nullptr, 0, 0, OUT, DIM_);

    // 7) out += gate * (H[e] * W2[e])  (atomicAdd scatter; ordered after 6 by stream)
    moe_gemm_kernel<1, true><<<dim3(DIM_ / 64, T_TOK / 128, NEXP_), 256, 0, stream>>>(
        H, (size_t)T_TOK * INTER_, INTER_,
        W2, nullptr, (size_t)INTER_ * DIM_, DIM_, INTER_,
        counts, T_TOK, tokenList, T_TOK, gateW,
        nullptr, 0, 0, OUT, DIM_);
}
